// KalmanNetWithKnownR_59322088292443
// MI455X (gfx1250) — compile-verified, matrix operand order checked by reading, not run on hardware
//
#include <hip/hip_runtime.h>
#include <hip/hip_bf16.h>

// ---------------------------------------------------------------------------
// KalmanNet recurrent filter for MI455X (gfx1250).
// Heavy math (gx = a@W_ih^T, gh = h@W_hh^T, [1024 x 2400 x 800] each) runs on
// v_wmma_f32_16x16x32_bf16. Weights converted fp32->bf16 once per call (L2-
// resident). 512 timesteps -> 2 kernels/step: gru_gemm (WMMA) + step_small
// (fused output-correction(t) + prep(t+1), all row-local -> one block/row).
// ---------------------------------------------------------------------------

#define STATE_DIM 8
#define OBS_DIM   4
#define HID       800
#define HID3      2400
#define BATCH     1024
#define SEQ       512

typedef __attribute__((ext_vector_type(16))) __bf16 v16bf;
typedef __attribute__((ext_vector_type(8)))  __bf16 v8bf;
typedef __attribute__((ext_vector_type(8)))  float  v8f;

// fp32 -> bf16, round-to-nearest-even.
__device__ __forceinline__ __bf16 f2bf(float f) {
    union { float f; unsigned u; } v; v.f = f;
    unsigned r = v.u + 0x7FFFu + ((v.u >> 16) & 1u);
    unsigned short h = (unsigned short)(r >> 16);
    return __builtin_bit_cast(__bf16, h);
}

__device__ __forceinline__ float sigmoidf_(float x) {
    return 1.0f / (1.0f + __expf(-x));
}

// Load one 16x32 bf16 WMMA operand fragment for this lane.
// Layout (ISA 7.12.2, 16-bit A 16x32): lanes 0-15 row=lane, VGPR0-3 K=kb..kb+7,
// VGPR4-7 K=kb+16..kb+23, kb = 0 (lanes 0-15) or 8 (lanes 16-31). B mirrors A
// with N in M's role; W stored [N,K] row-major is already K-contiguous.
__device__ __forceinline__ v16bf load_frag16(const __bf16* __restrict__ p) {
    v8bf lo = *reinterpret_cast<const v8bf*>(p);
    v8bf hi = *reinterpret_cast<const v8bf*>(p + 16);
    v16bf f;
#pragma unroll
    for (int i = 0; i < 8; ++i) { f[i] = lo[i]; f[8 + i] = hi[i]; }
    return f;
}

__device__ __forceinline__ v8f wmma_bf16(v16bf a, v16bf b, v8f c) {
    return __builtin_amdgcn_wmma_f32_16x16x32_bf16(false, a, false, b,
                                                   (short)0, c, false, false);
}

// ---------------------------------------------------------------------------
// Convert fp32 weights to bf16 (grid-stride).
// ---------------------------------------------------------------------------
__global__ __launch_bounds__(256)
void cvt_bf16_kernel(const float* __restrict__ src, __bf16* __restrict__ dst, int n) {
    int i = blockIdx.x * 256 + threadIdx.x;
    int stride = gridDim.x * 256;
    for (int j = i; j < n; j += stride) dst[j] = f2bf(src[j]);
}

// ---------------------------------------------------------------------------
// Zero-init hidden state (fp32 + bf16 shadow).
// ---------------------------------------------------------------------------
__global__ __launch_bounds__(256)
void init_kernel(float* __restrict__ h0, __bf16* __restrict__ h0b) {
    int i = blockIdx.x * 256 + threadIdx.x;
    int stride = gridDim.x * 256;
    __bf16 z = f2bf(0.f);
    for (int j = i; j < BATCH * HID; j += stride) { h0[j] = 0.f; h0b[j] = z; }
}

// ---------------------------------------------------------------------------
// gru_gemm: gx = a@W_ih^T, gh = h@W_hh^T for a 64(batch) x 32(hid) tile, then
// r/z/n gates and h_new = (1-z)*n + z*h. 8 waves:
//   waves 0-3 : gx for m-subtile (wave&3), 3 gates x 2 n-subtiles
//   waves 4-7 : gh for same tiles; crossed via LDS.
// K loop: 800 in steps of 32 -> v_wmma_f32_16x16x32_bf16.
// ---------------------------------------------------------------------------
__global__ __launch_bounds__(256)
void gru_gemm_kernel(const __bf16* __restrict__ a_bf,
                     const __bf16* __restrict__ h_bf,
                     const float*  __restrict__ h_prev,
                     const __bf16* __restrict__ Wih,
                     const __bf16* __restrict__ Whh,
                     const float* __restrict__ b_ih,
                     const float* __restrict__ b_hh,
                     float* __restrict__ h_next,
                     __bf16* __restrict__ h_next_bf) {
    const int tid  = threadIdx.x;
    const int wid  = tid >> 5;
    const int lane = tid & 31;
    const int side = wid >> 2;      // 0: gx (A = a), 1: gh (A = h)
    const int msub = wid & 3;
    const int m0 = blockIdx.x * 64 + msub * 16;
    const int n0 = blockIdx.y * 32;

    const __bf16* __restrict__ Amat = side ? h_bf : a_bf;
    const __bf16* __restrict__ Wmat = side ? Whh : Wih;

    const int r  = lane & 15;
    const int kb = (lane >> 4) << 3; // 0 or 8

    const __bf16* aptr = Amat + (size_t)(m0 + r) * HID + kb;
    const __bf16* wptr[3][2];
#pragma unroll
    for (int g = 0; g < 3; ++g)
#pragma unroll
        for (int s = 0; s < 2; ++s)
            wptr[g][s] = Wmat + (size_t)(g * HID + n0 + s * 16 + r) * HID + kb;

    v8f acc[3][2];
#pragma unroll
    for (int g = 0; g < 3; ++g)
#pragma unroll
        for (int s = 0; s < 2; ++s) acc[g][s] = (v8f)(0.0f);

    for (int kt = 0; kt < HID; kt += 32) {
        v16bf af = load_frag16(aptr + kt);
#pragma unroll
        for (int g = 0; g < 3; ++g) {
#pragma unroll
            for (int s = 0; s < 2; ++s) {
                v16bf bf = load_frag16(wptr[g][s] + kt);
                acc[g][s] = wmma_bf16(af, bf, acc[g][s]);
            }
        }
    }

    // Cross gh accumulators (waves 4-7) to gx waves (0-3) via LDS.
    __shared__ float lds_gh[4][3][2][256]; // 24 KB
    if (side == 1) {
#pragma unroll
        for (int g = 0; g < 3; ++g)
#pragma unroll
            for (int s = 0; s < 2; ++s)
#pragma unroll
                for (int v = 0; v < 8; ++v)
                    lds_gh[msub][g][s][v * 32 + lane] = acc[g][s][v];
    }
    __syncthreads();

    if (side == 0) {
        const int mrow_base = m0 + ((lane >> 4) << 3); // +0 or +8
#pragma unroll
        for (int s = 0; s < 2; ++s) {
            const int jc = n0 + s * 16 + (lane & 15);
            const float bir = b_ih[jc],            bhr = b_hh[jc];
            const float biz = b_ih[HID + jc],      bhz = b_hh[HID + jc];
            const float bin = b_ih[2 * HID + jc],  bhn = b_hh[2 * HID + jc];
#pragma unroll
            for (int v = 0; v < 8; ++v) {
                const int m = mrow_base + v;
                const float gxr = acc[0][s][v] + bir;
                const float gxz = acc[1][s][v] + biz;
                const float gxn = acc[2][s][v] + bin;
                const float ghr = lds_gh[msub][0][s][v * 32 + lane] + bhr;
                const float ghz = lds_gh[msub][1][s][v * 32 + lane] + bhz;
                const float ghn = lds_gh[msub][2][s][v * 32 + lane] + bhn;
                const float rg = sigmoidf_(gxr + ghr);
                const float zg = sigmoidf_(gxz + ghz);
                const float ng = tanhf(gxn + rg * ghn);
                const float hp = h_prev[(size_t)m * HID + jc];
                const float hn = (1.0f - zg) * ng + zg * hp;
                h_next[(size_t)m * HID + jc]    = hn;
                h_next_bf[(size_t)m * HID + jc] = f2bf(hn);
            }
        }
    }
}

// ---------------------------------------------------------------------------
// step_small: fused per-batch-row kernel (one block per row).
//   C-part (if t >= 0): K = h(t)@W_out^T + b_out, corr = K . innov(t),
//     x_hat(t) = x_pred(t) + corr, write output row.
//   A-part (if t+1 < SEQ): x_pred(t+1) = F x_hat, innov(t+1) = y(t+1) - H x_pred,
//     nn_in = [norm(dx), norm(innov)], a(t+1) = relu(W_in nn_in + b_in) -> bf16.
// All dependencies are row-local, so x_hat/dx stay in LDS (no global state).
// ---------------------------------------------------------------------------
__global__ __launch_bounds__(256)
void step_small_kernel(const float* __restrict__ y_seq, const float* __restrict__ Fm,
                       const float* __restrict__ Hm, const float* __restrict__ W_in,
                       const float* __restrict__ b_in, const float* __restrict__ W_out,
                       const float* __restrict__ b_out,
                       const float* __restrict__ h_new,   // fp32 h at time t (unused if t<0)
                       float* __restrict__ x_pred,        // [B,8] state: read(t), write(t+1)
                       float* __restrict__ innov,         // [B,4] state: read(t), write(t+1)
                       __bf16* __restrict__ a_bf,         // [B,800] write for t+1
                       float* __restrict__ out, int t) {
    const int b = blockIdx.x;
    const int tid = threadIdx.x;
    __shared__ float red[32][8];                  // partial dots -> K values
    __shared__ float xh[STATE_DIM];               // x_hat(t)
    __shared__ float dxs[STATE_DIM];              // corr(t)
    __shared__ float inn_c[OBS_DIM];              // innov(t)
    __shared__ float xp_n[STATE_DIM];             // x_pred(t+1)
    __shared__ float inn_n[OBS_DIM];              // innov(t+1)
    __shared__ float nn[STATE_DIM + OBS_DIM];     // normalized NN input

    if (t >= 0) {
        if (tid < OBS_DIM) inn_c[tid] = innov[b * OBS_DIM + tid];
        __syncthreads();
        // K[s,o] dot products: 32 (s,o) groups x 8-way K split (100 elems each).
        const int so = tid >> 3;   // 0..31
        const int part = tid & 7;  // 0..7
        const float* hr = h_new + (size_t)b * HID;
        const float* wr = W_out + (size_t)so * HID;
        float psum = 0.f;
        const int j0 = part * 100;
        for (int j = j0; j < j0 + 100; ++j) psum += hr[j] * wr[j];
        red[so][part] = psum;
        __syncthreads();
        if (tid < 32) {
            float kso = b_out[tid];
#pragma unroll
            for (int p = 0; p < 8; ++p) kso += red[tid][p];
            red[tid][0] = kso;  // K value
        }
        __syncthreads();
        if (tid < STATE_DIM) {
            float corr = 0.f;
#pragma unroll
            for (int o = 0; o < OBS_DIM; ++o) corr += red[tid * OBS_DIM + o][0] * inn_c[o];
            dxs[tid] = corr;
            const float x = x_pred[b * STATE_DIM + tid] + corr;
            xh[tid] = x;
            out[((size_t)b * SEQ + t) * STATE_DIM + tid] = x;
        }
        __syncthreads();
    } else {
        if (tid < STATE_DIM) { xh[tid] = 0.f; dxs[tid] = 0.f; }
        __syncthreads();
    }

    // ---- A-part for time t+1 ----
    const int tn = t + 1;
    if (tn >= SEQ) return;

    if (tid < STATE_DIM) {
        float s = 0.f;
#pragma unroll
        for (int k = 0; k < STATE_DIM; ++k) s += Fm[tid * STATE_DIM + k] * xh[k];
        xp_n[tid] = s;
        x_pred[b * STATE_DIM + tid] = s;
    }
    __syncthreads();
    if (tid < OBS_DIM) {
        float s = 0.f;
#pragma unroll
        for (int k = 0; k < STATE_DIM; ++k) s += Hm[tid * STATE_DIM + k] * xp_n[k];
        float iv = y_seq[((size_t)b * SEQ + tn) * OBS_DIM + tid] - s;
        inn_n[tid] = iv;
        innov[b * OBS_DIM + tid] = iv;
    }
    __syncthreads();
    if (tid == 0) {
        float nd = 0.f;
#pragma unroll
        for (int k = 0; k < STATE_DIM; ++k) nd += dxs[k] * dxs[k];
        nd = fmaxf(__fsqrt_rn(nd), 1e-12f);
#pragma unroll
        for (int k = 0; k < STATE_DIM; ++k) nn[k] = dxs[k] / nd;
        float ni = 0.f;
#pragma unroll
        for (int k = 0; k < OBS_DIM; ++k) ni += inn_n[k] * inn_n[k];
        ni = fmaxf(__fsqrt_rn(ni), 1e-12f);
#pragma unroll
        for (int k = 0; k < OBS_DIM; ++k) nn[STATE_DIM + k] = inn_n[k] / ni;
    }
    __syncthreads();
    const int IN = STATE_DIM + OBS_DIM; // 12
    for (int j = tid; j < HID; j += 256) {
        float s = b_in[j];
#pragma unroll
        for (int k = 0; k < IN; ++k) s += nn[k] * W_in[j * IN + k];
        a_bf[b * HID + j] = f2bf(fmaxf(s, 0.f));
    }
}

// ---------------------------------------------------------------------------
// Host-side launch sequencing: 1 init pass, then 2 kernels per timestep.
// ---------------------------------------------------------------------------
extern "C" void kernel_launch(void* const* d_in, const int* in_sizes, int n_in,
                              void* d_out, int out_size, void* d_ws, size_t ws_size,
                              hipStream_t stream) {
    const float* y_seq = (const float*)d_in[0];
    const float* Fm    = (const float*)d_in[1];
    const float* Hm    = (const float*)d_in[2];
    const float* W_in  = (const float*)d_in[3];
    const float* b_in  = (const float*)d_in[4];
    const float* W_ih  = (const float*)d_in[5];
    const float* b_ih  = (const float*)d_in[6];
    const float* W_hh  = (const float*)d_in[7];
    const float* b_hh  = (const float*)d_in[8];
    const float* W_out = (const float*)d_in[9];
    const float* b_out = (const float*)d_in[10];
    float* out = (float*)d_out;

    char* ws = (char*)d_ws;
    size_t off = 0;
    auto take = [&](size_t bytes) -> char* {
        char* p = ws + off;
        off = (off + bytes + 255) & ~(size_t)255;
        return p;
    };
    __bf16* Wih_bf  = (__bf16*)take((size_t)HID3 * HID * 2);
    __bf16* Whh_bf  = (__bf16*)take((size_t)HID3 * HID * 2);
    __bf16* a_bf    = (__bf16*)take((size_t)BATCH * HID * 2);
    float*  h_f32_0 = (float*) take((size_t)BATCH * HID * 4);
    float*  h_f32_1 = (float*) take((size_t)BATCH * HID * 4);
    __bf16* h_bf_0  = (__bf16*)take((size_t)BATCH * HID * 2);
    __bf16* h_bf_1  = (__bf16*)take((size_t)BATCH * HID * 2);
    float*  x_pred  = (float*) take((size_t)BATCH * STATE_DIM * 4);
    float*  innov   = (float*) take((size_t)BATCH * OBS_DIM * 4);
    (void)ws_size; (void)in_sizes; (void)n_in; (void)out_size;

    cvt_bf16_kernel<<<2048, 256, 0, stream>>>(W_ih, Wih_bf, HID3 * HID);
    cvt_bf16_kernel<<<2048, 256, 0, stream>>>(W_hh, Whh_bf, HID3 * HID);
    init_kernel<<<1024, 256, 0, stream>>>(h_f32_0, h_bf_0);
    // Bootstrap: A-part only (x0 = dx0 = 0) producing x_pred(0), innov(0), a(0).
    step_small_kernel<<<BATCH, 256, 0, stream>>>(y_seq, Fm, Hm, W_in, b_in,
                                                 W_out, b_out, h_f32_0,
                                                 x_pred, innov, a_bf, out, -1);

    float*  h_f32[2] = { h_f32_0, h_f32_1 };
    __bf16* h_bf[2]  = { h_bf_0,  h_bf_1  };
    int cur = 0;
    for (int t = 0; t < SEQ; ++t) {
        gru_gemm_kernel<<<dim3(BATCH / 64, HID / 32), 256, 0, stream>>>(
            a_bf, h_bf[cur], h_f32[cur], Wih_bf, Whh_bf, b_ih, b_hh,
            h_f32[cur ^ 1], h_bf[cur ^ 1]);
        step_small_kernel<<<BATCH, 256, 0, stream>>>(y_seq, Fm, Hm, W_in, b_in,
                                                     W_out, b_out, h_f32[cur ^ 1],
                                                     x_pred, innov, a_bf, out, t);
        cur ^= 1;
    }
}
